// Model_773094113452
// MI455X (gfx1250) — matrix operand order repaired and checked by reading, not compiled, not run on hardware
//
#include <hip/hip_runtime.h>
#include <hip/hip_bf16.h>
#include <stdint.h>

// ---------------------------------------------------------------------------
// MI455X / gfx1250 implementation.
//  - All four sparse convs + the 64->512 linear run on v_wmma_f32_16x16x32_bf16
//    (bf16 inputs, fp32 accumulation). One wave32 owns a 16x64 output tile.
//  - Conv weights are device-packed into the per-lane WMMA fragment layout,
//    then double-buffered through LDS with global_load_async_to_lds_b128 so
//    all 8 waves of a block share one 8KB B-fragment fetch per k-offset
//    (cuts B traffic 8x; A gathers are the irreducible L2-resident stream).
//  - Classifier (64->255) + log-softmax + CE is fused (no [M,255] tensor).
// ---------------------------------------------------------------------------

typedef __attribute__((ext_vector_type(16))) __bf16 v16bf_t;
typedef __attribute__((ext_vector_type(8)))  __bf16 v8bf_t;
typedef __attribute__((ext_vector_type(8)))  float  v8f_t;

#define KOFF 27

static __device__ __forceinline__ v16bf_t cat8(v8bf_t lo, v8bf_t hi) {
  return __builtin_shufflevector(lo, hi, 0,1,2,3,4,5,6,7,8,9,10,11,12,13,14,15);
}

static __device__ __forceinline__ uint32_t lds_addr_of(const void* p) {
  // generic -> LDS(addrspace 3) cast; ptrtoint yields the 32-bit LDS offset
  return (uint32_t)(size_t)(__attribute__((address_space(3))) const void*)p;
}

// ---- weight packing: conv weights [27][64][64] f32 -> per-lane bf16 frags ---
// layout: (((k*2 + c)*4 + t)*32 + lane)*16 + e
//   B frag: K = c*32 + (e>>3)*16 + (lane>>4)*8 + (e&7),  N = t*16 + (lane&15)
__global__ void pack_conv_w(const float* __restrict__ W, __bf16* __restrict__ P) {
  int id = blockIdx.x * blockDim.x + threadIdx.x;
  if (id >= KOFF * 4096) return;
  int e    = id & 15;
  int lane = (id >> 4) & 31;
  int t    = (id >> 9) & 3;
  int c    = (id >> 11) & 1;
  int k    = id >> 12;
  int kk   = c * 32 + ((e >> 3) << 4) + ((lane >> 4) << 3) + (e & 7);
  int col  = t * 16 + (lane & 15);
  P[id] = (__bf16)W[((size_t)k * 64 + kk) * 64 + col];
}

// ---- Wl0 [64][512] f32 -> packed bf16 frags: ((c*32+tg)*32 + lane)*16 + e ---
__global__ void pack_lin_w(const float* __restrict__ W, __bf16* __restrict__ P) {
  int id = blockIdx.x * blockDim.x + threadIdx.x;
  if (id >= 32768) return;
  int e    = id & 15;
  int lane = (id >> 4) & 31;
  int tg   = (id >> 9) & 31;
  int c    = id >> 14;
  int kk   = c * 32 + ((e >> 3) << 4) + ((lane >> 4) << 3) + (e & 7);
  int col  = tg * 16 + (lane & 15);
  P[id] = (__bf16)W[(size_t)kk * 512 + col];
}

// ---- Wl1 [64][255] -> padded [64][256] f32 (aligned column loads) ----------
__global__ void pack_wl1(const float* __restrict__ W, float* __restrict__ P) {
  int id = blockIdx.x * blockDim.x + threadIdx.x;
  if (id >= 64 * 256) return;
  int col = id & 255;
  int d   = id >> 8;
  P[id] = (col < 255) ? W[(size_t)d * 255 + col] : 0.0f;
}

// ---- dec: x = PReLU([F | bins1] @ W_dec + b_dec); writes f32 + bf16 --------
__global__ void dec_kernel(const float* __restrict__ F, const float* __restrict__ bins1,
                           const float* __restrict__ Wd, const float* __restrict__ bd,
                           const float* __restrict__ a_ptr,
                           float* __restrict__ x_f, __bf16* __restrict__ x_bf, int n) {
  int id = blockIdx.x * blockDim.x + threadIdx.x;
  if (id >= n * 64) return;
  int row = id >> 6, col = id & 63;
  float acc = bd[col];
  const float* fr = F + (size_t)row * 64;
  #pragma unroll 8
  for (int d = 0; d < 64; ++d) acc += fr[d] * Wd[d * 64 + col];
  const float* br = bins1 + (size_t)row * 8;
  #pragma unroll
  for (int d = 0; d < 8; ++d) acc += br[d] * Wd[(64 + d) * 64 + col];
  float a = a_ptr[0];
  acc = (acc >= 0.0f) ? acc : a * acc;
  x_f[id]  = acc;
  x_bf[id] = (__bf16)acc;
}

// ---- masked-gather sparse conv, WMMA bf16 -> f32, fused bias/res/PReLU -----
// 8 waves/block, one 16-row tile each; B fragments double-buffered via
// async global->LDS copies shared by the whole block.
__global__ void __launch_bounds__(256)
spconv_wmma(const __bf16* __restrict__ xin, const int* __restrict__ nbr,
            const unsigned char* __restrict__ mask, const __bf16* __restrict__ pW,
            const float* __restrict__ bias, const float* __restrict__ res,
            const float* __restrict__ slope, float* __restrict__ out_f,
            __bf16* __restrict__ out_bf, int n) {
  __shared__ __bf16 Bbuf[2][4096];            // 2 x 8KB B-fragment buffers

  int tid  = threadIdx.x;
  int lane = tid & 31;
  int wv   = tid >> 5;
  int tile = blockIdx.x * 8 + wv;
  int base = tile * 16;
  bool active = (base < n);                   // wave-uniform

  int m    = lane & 15;
  int half = lane >> 4;
  int r    = active ? (base + m) : 0;

  const int*           nrow = nbr  + (size_t)r * KOFF;
  const unsigned char* mrow = mask + (size_t)r * KOFF;

  uint32_t lds0 = lds_addr_of(&Bbuf[0][0]) + (uint32_t)tid * 32u;
  uint32_t lds1 = lds_addr_of(&Bbuf[1][0]) + (uint32_t)tid * 32u;
  uint64_t gsrc = (uint64_t)(size_t)pW + (uint64_t)tid * 32u;

  // prologue: stage k=0 fragments into buffer 0 (each thread: 2 x b128 = 32B)
  asm volatile("global_load_async_to_lds_b128 %0, %1, off\n\t"
               "global_load_async_to_lds_b128 %0, %1, off offset:16"
               :: "v"(lds0), "v"(gsrc) : "memory");

  v8f_t acc[4] = {};

  for (int k = 0; k < KOFF; ++k) {
    if (k + 1 < KOFF) {
      uint32_t ldsn = ((k + 1) & 1) ? lds1 : lds0;
      uint64_t gn   = gsrc + (uint64_t)(k + 1) * 8192u;
      asm volatile("global_load_async_to_lds_b128 %0, %1, off\n\t"
                   "global_load_async_to_lds_b128 %0, %1, off offset:16"
                   :: "v"(ldsn), "v"(gn) : "memory");
      // async loads complete in order: <=2 outstanding => batch k has landed
      asm volatile("s_wait_asynccnt 0x2" ::: "memory");
    } else {
      asm volatile("s_wait_asynccnt 0x0" ::: "memory");
    }
    __syncthreads();                          // publish batch k to all waves

    if (active) {
      int  nb = nrow[k];
      bool mk = (mrow[k] != 0);
      const __bf16* xr = xin + (size_t)nb * 64;
      // A fragments per ISA 16-bit 16x32 layout
      v8bf_t a0l = *(const v8bf_t*)(xr +       (half << 3));
      v8bf_t a0h = *(const v8bf_t*)(xr + 16 +  (half << 3));
      v8bf_t a1l = *(const v8bf_t*)(xr + 32 +  (half << 3));
      v8bf_t a1h = *(const v8bf_t*)(xr + 48 +  (half << 3));
      if (!mk) {
        v8bf_t z = {};
        a0l = z; a0h = z; a1l = z; a1h = z;
      }
      v16bf_t A0 = cat8(a0l, a0h);
      v16bf_t A1 = cat8(a1l, a1h);

      const __bf16* bk = &Bbuf[k & 1][lane * 16];
      #pragma unroll
      for (int t = 0; t < 4; ++t) {
        v16bf_t B0 = *(const v16bf_t*)(bk + t * 512);
        v16bf_t B1 = *(const v16bf_t*)(bk + 2048 + t * 512);
        acc[t] = __builtin_amdgcn_wmma_f32_16x16x32_bf16(false, A0, false, B0,
                                                         (short)0, acc[t], false, false);
        acc[t] = __builtin_amdgcn_wmma_f32_16x16x32_bf16(false, A1, false, B1,
                                                         (short)0, acc[t], false, false);
      }
    }
    __syncthreads();                          // protect buffer before overwrite
  }

  if (active) {
    float a = slope ? slope[0] : 0.0f;
    #pragma unroll
    for (int t = 0; t < 4; ++t) {
      int   col = t * 16 + m;
      float bv  = bias[col];
      #pragma unroll
      for (int v = 0; v < 8; ++v) {
        int    row = base + v + half * 8;      // C/D layout: M = v + 8*half
        size_t idx = (size_t)row * 64 + col;
        float  y   = acc[t][v] + bv;
        if (res)    y += res[idx];
        if (slope)  y  = (y >= 0.0f) ? y : a * y;
        if (out_f)  out_f[idx]  = y;
        if (out_bf) out_bf[idx] = (__bf16)y;
      }
    }
  }
}

// ---- p = u @ Wl0 + bl0 : [n,64] x [64,512], WMMA, bf16 out -----------------
__global__ void __launch_bounds__(256)
gemm512_wmma(const __bf16* __restrict__ xin, const __bf16* __restrict__ pW,
             const float* __restrict__ bias, __bf16* __restrict__ out, int n) {
  int gid  = blockIdx.x * blockDim.x + threadIdx.x;
  int wid  = gid >> 5;
  int lane = gid & 31;
  int tile = wid >> 3;          // 16-row tile
  int g    = wid & 7;           // 64-col group
  int base = tile * 16;
  if (base >= n) return;
  int m    = lane & 15;
  int half = lane >> 4;

  const __bf16* xr = xin + (size_t)(base + m) * 64;
  v8bf_t a0l = *(const v8bf_t*)(xr +      (half << 3));
  v8bf_t a0h = *(const v8bf_t*)(xr + 16 + (half << 3));
  v8bf_t a1l = *(const v8bf_t*)(xr + 32 + (half << 3));
  v8bf_t a1h = *(const v8bf_t*)(xr + 48 + (half << 3));
  v16bf_t A0 = cat8(a0l, a0h);
  v16bf_t A1 = cat8(a1l, a1h);

  v8f_t acc[4] = {};
  #pragma unroll
  for (int t = 0; t < 4; ++t) {
    int tg = g * 4 + t;
    v16bf_t B0 = *(const v16bf_t*)(pW + ((size_t)(tg)      * 32 + lane) * 16);
    v16bf_t B1 = *(const v16bf_t*)(pW + ((size_t)(32 + tg) * 32 + lane) * 16);
    acc[t] = __builtin_amdgcn_wmma_f32_16x16x32_bf16(false, A0, false, B0,
                                                     (short)0, acc[t], false, false);
    acc[t] = __builtin_amdgcn_wmma_f32_16x16x32_bf16(false, A1, false, B1,
                                                     (short)0, acc[t], false, false);
  }

  #pragma unroll
  for (int t = 0; t < 4; ++t) {
    int col = g * 64 + t * 16 + m;
    #pragma unroll
    for (int v = 0; v < 8; ++v) {
      int row = base + v + half * 8;
      out[(size_t)row * 512 + col] = (__bf16)(acc[t][v] + bias[col]);
    }
  }
}

// ---- tmp[j] = p.reshape(N*8,64)[sel_idx[j]] --------------------------------
__global__ void gather_kernel(const __bf16* __restrict__ p, const int* __restrict__ sel,
                              __bf16* __restrict__ tmp, int mtot) {
  int id = blockIdx.x * blockDim.x + threadIdx.x;
  if (id >= mtot * 8) return;
  int j = id >> 3, chunk = id & 7;
  *(v8bf_t*)(tmp + (size_t)j * 64 + chunk * 8) =
      *(const v8bf_t*)(p + (size_t)sel[j] * 64 + chunk * 8);
}

// ---- fused classifier + log-softmax + CE (one wave per fine point) ---------
__global__ void __launch_bounds__(256)
ce_kernel(const __bf16* __restrict__ q, const float* __restrict__ pWl1,
          const float* __restrict__ bl1, const unsigned char* __restrict__ bins0,
          float* __restrict__ loss_acc, int mtot) {
  __shared__ float lq[8][64];
  __shared__ float lce[8];
  int lane = threadIdx.x & 31;
  int w    = threadIdx.x >> 5;
  int j    = blockIdx.x * 8 + w;
  if (j < mtot) {
    lq[w][lane * 2]     = (float)q[(size_t)j * 64 + lane * 2];
    lq[w][lane * 2 + 1] = (float)q[(size_t)j * 64 + lane * 2 + 1];
  }
  if (lane == 0) lce[w] = 0.0f;
  __syncthreads();
  if (j < mtot) {
    int   colb = lane * 8;
    float acc[8];
    #pragma unroll
    for (int c = 0; c < 8; ++c) acc[c] = 0.0f;
    for (int d = 0; d < 64; ++d) {
      float qd = lq[w][d];
      const float* wr = pWl1 + d * 256 + colb;
      #pragma unroll
      for (int c = 0; c < 8; ++c) acc[c] += qd * wr[c];
    }
    float lmax = -1e30f;
    #pragma unroll
    for (int c = 0; c < 8; ++c) {
      int col = colb + c;
      acc[c] = (col < 255) ? (acc[c] + bl1[col]) : -1e30f;
      lmax = fmaxf(lmax, acc[c]);
    }
    for (int s = 16; s > 0; s >>= 1) lmax = fmaxf(lmax, __shfl_xor(lmax, s, 32));
    float lsum = 0.0f;
    #pragma unroll
    for (int c = 0; c < 8; ++c) lsum += __expf(acc[c] - lmax);
    for (int s = 16; s > 0; s >>= 1) lsum += __shfl_xor(lsum, s, 32);
    const unsigned char* b = bins0 + (size_t)j * 8;
    int oct = 0;
    #pragma unroll
    for (int t = 0; t < 8; ++t) oct += b[t] ? (1 << t) : 0;
    oct -= 1;
    if ((oct >> 3) == lane) {
      lce[w] = lmax + __logf(lsum) - acc[oct & 7];
    }
  }
  __syncthreads();
  if (threadIdx.x == 0) {
    float s = 0.0f;
    #pragma unroll
    for (int i = 0; i < 8; ++i) s += lce[i];
    atomicAdd(loss_acc, s);
  }
}

__global__ void finalize_kernel(const float* __restrict__ loss_acc,
                                float* __restrict__ out, float scale) {
  out[0] = loss_acc[0] * scale;
}

// ---------------------------------------------------------------------------
extern "C" void kernel_launch(void* const* d_in, const int* in_sizes, int n_in,
                              void* d_out, int out_size, void* d_ws, size_t ws_size,
                              hipStream_t stream) {
  const float*         cur_rec_F = (const float*)d_in[0];
  const float*         bins1_f   = (const float*)d_in[1];
  const unsigned char* bins0     = (const unsigned char*)d_in[2];
  const int*           sel_idx   = (const int*)d_in[3];
  const int*           nbr0      = (const int*)d_in[4];
  const unsigned char* mask0     = (const unsigned char*)d_in[5];
  const int*           nbr1      = (const int*)d_in[6];
  const unsigned char* mask1     = (const unsigned char*)d_in[7];
  const float* W_dec = (const float*)d_in[9];
  const float* b_dec = (const float*)d_in[10];
  const float* a_dec = (const float*)d_in[11];
  const float* Wb1   = (const float*)d_in[12];
  const float* bb1   = (const float*)d_in[13];
  const float* ab1   = (const float*)d_in[14];
  const float* Wb2   = (const float*)d_in[15];
  const float* bb2   = (const float*)d_in[16];
  const float* ab2   = (const float*)d_in[17];
  const float* Wp0   = (const float*)d_in[18];
  const float* bp0   = (const float*)d_in[19];
  const float* ap0   = (const float*)d_in[20];
  const float* Wl0   = (const float*)d_in[21];
  const float* bl0   = (const float*)d_in[22];
  const float* Wp1   = (const float*)d_in[23];
  const float* bp1   = (const float*)d_in[24];
  const float* ap1   = (const float*)d_in[25];
  const float* Wl1   = (const float*)d_in[26];
  const float* bl1   = (const float*)d_in[27];

  const int N = in_sizes[0] / 64;   // 150000
  const int M = in_sizes[3];        // 600000

  // ---- carve workspace (256B aligned chunks) ----
  char* wp = (char*)d_ws;
  auto carve = [&](size_t bytes) -> void* {
    void* r = (void*)wp;
    wp += (bytes + 255) & ~(size_t)255;
    return r;
  };
  __bf16* pWb1   = (__bf16*)carve((size_t)KOFF * 4096 * 2);
  __bf16* pWb2   = (__bf16*)carve((size_t)KOFF * 4096 * 2);
  __bf16* pWp0   = (__bf16*)carve((size_t)KOFF * 4096 * 2);
  __bf16* pWp1   = (__bf16*)carve((size_t)KOFF * 4096 * 2);
  __bf16* pWl0   = (__bf16*)carve((size_t)32768 * 2);
  float*  pWl1   = (float*)carve((size_t)64 * 256 * 4);
  float*  lossp  = (float*)carve(256);
  float*  x_f    = (float*)carve((size_t)N * 64 * 4);
  __bf16* x_bf   = (__bf16*)carve((size_t)N * 64 * 2);
  __bf16* h_bf   = (__bf16*)carve((size_t)N * 64 * 2);
  __bf16* rec_bf = (__bf16*)carve((size_t)N * 64 * 2);
  __bf16* u_bf   = (__bf16*)carve((size_t)N * 64 * 2);
  __bf16* p_bf   = (__bf16*)carve((size_t)N * 512 * 2);
  __bf16* tmp_bf = (__bf16*)carve((size_t)M * 64 * 2);
  __bf16* q_bf   = (__bf16*)carve((size_t)M * 64 * 2);

  float* rec_out = (float*)d_out;            // [N,64] then loss scalar

  hipMemsetAsync(lossp, 0, sizeof(float), stream);

  // ---- pack weights into WMMA fragment layout ----
  pack_conv_w<<<(KOFF * 4096 + 255) / 256, 256, 0, stream>>>(Wb1, pWb1);
  pack_conv_w<<<(KOFF * 4096 + 255) / 256, 256, 0, stream>>>(Wb2, pWb2);
  pack_conv_w<<<(KOFF * 4096 + 255) / 256, 256, 0, stream>>>(Wp0, pWp0);
  pack_conv_w<<<(KOFF * 4096 + 255) / 256, 256, 0, stream>>>(Wp1, pWp1);
  pack_lin_w <<<(32768 + 255) / 256, 256, 0, stream>>>(Wl0, pWl0);
  pack_wl1   <<<(64 * 256 + 255) / 256, 256, 0, stream>>>(Wl1, pWl1);

  // ---- dec: x = PReLU([F|bins1] @ W_dec + b) ----
  dec_kernel<<<((size_t)N * 64 + 255) / 256, 256, 0, stream>>>(
      cur_rec_F, bins1_f, W_dec, b_dec, a_dec, x_f, x_bf, N);

  const int tilesN = (N + 15) / 16;
  const int tilesM = (M + 15) / 16;
  const int gridN  = (tilesN + 7) / 8;
  const int gridM  = (tilesM + 7) / 8;

  // h = PReLU(conv(x, Wb1) + bb1)
  spconv_wmma<<<gridN, 256, 0, stream>>>(x_bf, nbr0, mask0, pWb1, bb1,
                                         nullptr, ab1, nullptr, h_bf, N);
  // rec = PReLU(conv(h, Wb2) + bb2 + x)  -> d_out + bf16 copy
  spconv_wmma<<<gridN, 256, 0, stream>>>(h_bf, nbr0, mask0, pWb2, bb2,
                                         x_f, ab2, rec_out, rec_bf, N);
  // u = PReLU(conv(rec, Wp0) + bp0)
  spconv_wmma<<<gridN, 256, 0, stream>>>(rec_bf, nbr0, mask0, pWp0, bp0,
                                         nullptr, ap0, nullptr, u_bf, N);
  // p = u @ Wl0 + bl0   [N,512]
  gemm512_wmma<<<(tilesN * 8 * 32 + 255) / 256, 256, 0, stream>>>(
      u_bf, pWl0, bl0, p_bf, N);
  // tmp = p.reshape(N*8,64)[sel_idx]
  gather_kernel<<<((size_t)M * 8 + 255) / 256, 256, 0, stream>>>(
      p_bf, sel_idx, tmp_bf, M);
  // q = PReLU(conv(tmp, Wp1) + bp1)
  spconv_wmma<<<gridM, 256, 0, stream>>>(tmp_bf, nbr1, mask1, pWp1, bp1,
                                         nullptr, ap1, nullptr, q_bf, M);
  // loss = CE(q @ Wl1 + bl1, labels) * log2(e) / M
  ce_kernel<<<(M + 7) / 8, 256, 0, stream>>>(q_bf, pWl1, bl1, bins0, lossp, M);
  finalize_kernel<<<1, 1, 0, stream>>>(lossp, rec_out + (size_t)N * 64,
                                       (float)(1.4426950408889634 / (double)M));
}
